// CrossModalTransformer_34050500723407
// MI455X (gfx1250) — compile-verified, matrix operand order checked by reading, not run on hardware
//
#include <hip/hip_runtime.h>
#include <math.h>

#define E 8

typedef __attribute__((ext_vector_type(16))) _Float16 v16h;
typedef __attribute__((ext_vector_type(8)))  _Float16 v8h;
typedef __attribute__((ext_vector_type(8)))  float    v8f;
typedef __attribute__((ext_vector_type(4)))  float    v4f;
typedef unsigned int u32;

__device__ __forceinline__ float pe_val(float pos, int oc) {
  int k = oc & ~1;
  float div = __expf((float)k * (-9.210340371976184f / 8.0f));
  float ang = pos * div;
  return (oc & 1) ? __cosf(ang) : __sinf(ang);
}

// ---------------------------------------------------------------------------
// One v_wmma_f32_16x16x32_f16 per 16-row chunk. K=8 real (zero-padded to 32).
// A (16-bit 16x32): lanes 0-15 row M=lane, halves 0..7 = K 0..7; lanes 16-31 zero.
// B (16-bit 32x16): lanes 0-15 col N=lane, halves 0..7 = K 0..7.
// D (f32 16x16):    lane -> col N=lane&15, VGPR r -> row M = r + 8*(lane>>4).
// Sources live in LDS as f16 rows (16B) -> A build = one ds_load_b128, no cvt.
// Bias is folded into the C accumulator (C[m][n] = bias[n]).
// ---------------------------------------------------------------------------

__device__ __forceinline__ v16h load_a_frag(const _Float16* src, int ch, int lane,
                                            u32 amask) {
  uint4 raw = *(const uint4*)(src + (ch * 16 + (lane & 15)) * 8);
  raw.x &= amask; raw.y &= amask; raw.z &= amask; raw.w &= amask;
  union { uint4 u; v8h h; } cv;
  cv.u = raw;
  v16h af;
  #pragma unroll
  for (int k = 0; k < 8; ++k) af[k] = cv.h[k];
  #pragma unroll
  for (int k = 8; k < 16; ++k) af[k] = (_Float16)0.0f;
  return af;
}

// Transposed-output projection. DUAL packs two 8x8 weight blocks into B's 16
// columns -> one WMMA yields both K and V. Writeback: 2 contiguous b128 stores.
template <bool DUAL>
__device__ __forceinline__ void wave_proj_t(const _Float16* src,
                                            const float* W1, const float* b1,
                                            const float* W2, const float* b2,
                                            float* d1T, float* d2T, int ld,
                                            int lane, int c0, int c1) {
  const int n = lane & 15;
  const int half = lane >> 4;
  const u32 amask = (lane < 16) ? 0xFFFFFFFFu : 0u;
  v16h bf = {};
  float bn = 0.0f;
  float* dcol = nullptr;
  if (n < 8) { bn = b1[n]; dcol = d1T + n * ld; }
  else if (DUAL) { bn = b2[n - 8]; dcol = d2T + (n - 8) * ld; }
  if (lane < 16 && (n < 8 || DUAL)) {
    const float* wrow = (n < 8) ? (W1 + n * 8) : (W2 + (n - 8) * 8);
    #pragma unroll
    for (int k = 0; k < 8; ++k) bf[k] = (_Float16)wrow[k];
  }
  for (int ch = c0; ch < c1; ++ch) {
    v16h af = load_a_frag(src, ch, lane, amask);
    v8f c = {bn, bn, bn, bn, bn, bn, bn, bn};
    c = __builtin_amdgcn_wmma_f32_16x16x32_f16(false, af, false, bf, (short)0, c,
                                               false, false);
    if (DUAL || n < 8) {
      const int rb = ch * 16 + 8 * half;
      *(v4f*)(dcol + rb)     = (v4f){c[0], c[1], c[2], c[3]};
      *(v4f*)(dcol + rb + 4) = (v4f){c[4], c[5], c[6], c[7]};
    }
  }
}

// Row-major f32 output projection (result consumed row-wise by layernorm/fc1).
__device__ __forceinline__ void wave_proj_rm(const _Float16* src, const float* W,
                                             const float* bias, float* dst,
                                             int lane, int c0, int c1) {
  const int n = lane & 15;
  const int half = lane >> 4;
  const u32 amask = (lane < 16) ? 0xFFFFFFFFu : 0u;
  v16h bf = {};
  if (lane < 16 && n < 8) {
    #pragma unroll
    for (int k = 0; k < 8; ++k) bf[k] = (_Float16)W[n * 8 + k];
  }
  const float bn = (n < 8) ? bias[n] : 0.0f;
  for (int ch = c0; ch < c1; ++ch) {
    v16h af = load_a_frag(src, ch, lane, amask);
    v8f c = {bn, bn, bn, bn, bn, bn, bn, bn};
    c = __builtin_amdgcn_wmma_f32_16x16x32_f16(false, af, false, bf, (short)0, c,
                                               false, false);
    if (n < 8) {
      const int rb = ch * 16 + 8 * half;
      #pragma unroll
      for (int r = 0; r < 8; ++r) dst[(rb + r) * 8 + n] = c[r];
    }
  }
}

// Per-lane softmax attention row (head_dim=1, scale=1). Transposed K/V allow
// contiguous b128 reads along j. Row result converted to f16 (next A-source).
__device__ __forceinline__ void attn_rows(int i, const float* qpT,
                                          const float* kpT, const float* vpT,
                                          int ldk, int Lk, _Float16* ob) {
  float o[8];
  for (int h = 0; h < 8; ++h) {
    const float qv = qpT[h * 32 + i];
    const float* kc = kpT + h * ldk;
    const float* vc = vpT + h * ldk;
    float m = -3.4e38f;
    int j = 0;
    for (; j + 3 < Lk; j += 4) {
      v4f k4 = *(const v4f*)(kc + j);
      m = fmaxf(m, fmaxf(fmaxf(qv * k4[0], qv * k4[1]),
                         fmaxf(qv * k4[2], qv * k4[3])));
    }
    for (; j < Lk; ++j) m = fmaxf(m, qv * kc[j]);
    float s = 0.0f, acc = 0.0f;
    j = 0;
    for (; j + 3 < Lk; j += 4) {
      v4f k4 = *(const v4f*)(kc + j);
      v4f vv4 = *(const v4f*)(vc + j);
      #pragma unroll
      for (int u = 0; u < 4; ++u) {
        float ev = __expf(qv * k4[u] - m);
        s += ev;
        acc += ev * vv4[u];
      }
    }
    for (; j < Lk; ++j) {
      float ev = __expf(qv * kc[j] - m);
      s += ev;
      acc += ev * vc[j];
    }
    o[h] = acc / s;
  }
  union { uint4 u; v8h h; } cv;
  #pragma unroll
  for (int h = 0; h < 8; ++h) cv.h[h] = (_Float16)o[h];
  *(uint4*)(ob + i * 8) = cv.u;
}

// Full MHA by one wave. scr (1024 f32): [0..255]=qpT[8][32], [256..511]=kpT,
// [512..767]=vpT, [768..1023]=row-major f32 result. ob = f16 [32][8] scratch.
__device__ void wave_mha(int lane, const _Float16* q, const _Float16* kv, int Lk,
                         const float* wIn, const float* bIn,
                         const float* wOut, const float* bOut,
                         float* scr, _Float16* ob) {
  wave_proj_t<false>(q, wIn, bIn, nullptr, nullptr, scr, nullptr, 32, lane, 0, 2);
  wave_proj_t<true>(kv, wIn + 64, bIn + 8, wIn + 128, bIn + 16,
                    scr + 256, scr + 512, 32, lane, 0, 2);
  attn_rows(lane, scr, scr + 256, scr + 512, 32, Lk, ob);
  wave_proj_rm(ob, wOut, bOut, scr + 768, lane, 0, 2);
}

__global__ __launch_bounds__(256) void cmt_kernel(
    const float* __restrict__ eeg, const float* __restrict__ pupil,
    const float* __restrict__ speech, const float* __restrict__ action,
    const float* __restrict__ location, const float* __restrict__ tgt,
    const float* __restrict__ eeg_w, const float* __restrict__ eeg_b,
    const float* __restrict__ psa_w, const float* __restrict__ psa_b,
    const float* __restrict__ loc_w, const float* __restrict__ loc_b,
    const float* __restrict__ tgt_w, const float* __restrict__ tgt_b,
    const float* __restrict__ ng, const float* __restrict__ nb,
    const float* __restrict__ cin_w, const float* __restrict__ cin_b,
    const float* __restrict__ cout_w, const float* __restrict__ cout_b,
    const float* __restrict__ sfin_w, const float* __restrict__ sfin_b,
    const float* __restrict__ sfout_w, const float* __restrict__ sfout_b,
    const float* __restrict__ oin_w, const float* __restrict__ oin_b,
    const float* __restrict__ oout_w, const float* __restrict__ oout_b,
    const float* __restrict__ fc1_w, const float* __restrict__ fc1_b,
    float* __restrict__ out) {
  __shared__ _Float16 s_seq[5][32][E];   // f16 A-sources: e(30), p, s, a, l
  __shared__ _Float16 s_t[32][E];
  __shared__ _Float16 s_concat[128][E];  // 126 rows used
  __shared__ _Float16 s_ob[8][32][E];    // per-wave attention-output (A-source)
  __shared__ float s_acc[4][32][E];
  __shared__ float s_scr[8][1024];       // per-wave qpT/kpT/vpT/result
  __shared__ float s_kpTf[8][128];       // final MHA K, transposed [head][pos]
  __shared__ float s_vpTf[8][128];
  __shared__ float s_tout[32][E];
  __shared__ float s_fc[96];

  const int b    = blockIdx.x;
  const int tid  = threadIdx.x;
  const int lane = tid & 31;
  const int wave = tid >> 5;
  const int Lseq[5] = {30, 32, 32, 32, 32};

  __builtin_prefetch(fc1_w + (tid << 6), 0, 1);   // global_prefetch_b8

  // ---------------- Phase 0: convs + positional encoding into LDS ----------
  for (int idx = tid; idx < 4 * 32 * E; idx += 256) ((float*)s_acc)[idx] = 0.0f;

  if (tid < 240) {  // eeg conv2d (2,20,118) -> 30x8, kernel (20,2), stride (1,4)
    int pos = tid >> 3, oc = tid & 7;
    const float* ep = eeg + (size_t)b * (2 * 20 * 118);
    float sum = eeg_b[oc];
    for (int ic = 0; ic < 2; ++ic) {
      #pragma unroll 4
      for (int kh = 0; kh < 20; ++kh) {
        const float* row = ep + (ic * 20 + kh) * 118 + pos * 4;
        const float* wr  = eeg_w + ((oc * 2 + ic) * 20 + kh) * 2;
        sum += row[0] * wr[0] + row[1] * wr[1];
      }
    }
    s_seq[0][pos][oc] = (_Float16)(sum + pe_val(30.0f, oc));
  }

  for (int idx = tid; idx < 5 * 32 * E; idx += 256) {  // p/s/a/l/tgt conv1d k1p1
    int sSel = idx >> 8;
    int pos  = (idx >> 3) & 31;
    int oc   = idx & 7;
    float v;
    _Float16* dst;
    if (sSel < 3) {
      const float* x = (sSel == 0 ? pupil : (sSel == 1 ? speech : action)) + (size_t)b * 60;
      v = psa_b[oc];
      if (pos >= 1 && pos <= 30)
        v += psa_w[oc * 2 + 0] * x[pos - 1] + psa_w[oc * 2 + 1] * x[30 + pos - 1];
      dst = &s_seq[1 + sSel][pos][oc];
    } else if (sSel == 3) {
      const float* x = location + (size_t)b * 90;
      v = loc_b[oc];
      if (pos >= 1 && pos <= 30)
        #pragma unroll
        for (int ic = 0; ic < 3; ++ic) v += loc_w[oc * 3 + ic] * x[ic * 30 + pos - 1];
      dst = &s_seq[4][pos][oc];
    } else {
      const float* x = tgt + (size_t)b * 30;
      v = tgt_b[oc];
      if (pos >= 1 && pos <= 30) v += tgt_w[oc] * x[pos - 1];
      dst = &s_t[pos][oc];
    }
    *dst = (_Float16)(v + pe_val(32.0f, oc));
  }
  __syncthreads();

  // ---------------- Phase 1: 18 MHAs, one task per wave --------------------
  const int TQ[18] = {0,0,0,0,0, 1,1,1,1,1, 2,2,2,2, 3,3,3,3};
  const int TK[18] = {1,3,4,0,2, 0,3,4,1,2, 0,1,3,4, 0,1,3,2};
  const int TS[18] = {0,0,0,1,0, 0,0,0,1,0, 0,0,0,0, 0,0,1,0};
  const int TA[18] = {0,0,0,0,0, 1,1,1,1,1, 2,2,2,2, 3,3,3,3};
  const int TB[18] = {-1,-1,-1,-1,-1, -1,-1,-1,-1,-1, -1,-1,-1,3, -1,-1,-1,-1};

  for (int t = wave; t < 18; t += 8) {
    const float* wIn  = TS[t] ? sfin_w  : cin_w;
    const float* bIn  = TS[t] ? sfin_b  : cin_b;
    const float* wOut = TS[t] ? sfout_w : cout_w;
    const float* bOut = TS[t] ? sfout_b : cout_b;
    int Lq = Lseq[TQ[t]], Lk = Lseq[TK[t]];
    float* scr = s_scr[wave];
    wave_mha(lane, &s_seq[TQ[t]][0][0], &s_seq[TK[t]][0][0], Lk,
             wIn, bIn, wOut, bOut, scr, &s_ob[wave][0][0]);
    // layernorm + accumulate (ds_add_f32 atomics across waves)
    int i = lane;
    if (i < Lq) {
      const float* row = scr + 768 + i * 8;
      float mu = 0.0f;
      #pragma unroll
      for (int c = 0; c < E; ++c) mu += row[c];
      mu *= 0.125f;
      float var = 0.0f;
      #pragma unroll
      for (int c = 0; c < E; ++c) { float d = row[c] - mu; var += d * d; }
      var *= 0.125f;
      float rinv = rsqrtf(var + 1e-5f);
      #pragma unroll
      for (int c = 0; c < E; ++c) {
        float v = (row[c] - mu) * rinv * ng[c] + nb[c];
        atomicAdd(&s_acc[TA[t]][i][c], v);
        if (TB[t] >= 0) atomicAdd(&s_acc[TB[t]][i][c], v);
      }
    }
  }
  __syncthreads();

  // ---------------- Phase 2: concat (f16) + final MHA ----------------------
  for (int idx = tid; idx < 126 * E; idx += 256) {  // [eeg30|pupil32|action32|speech32]
    int row = idx >> 3, c = idx & 7;
    int ai, r2;
    if (row < 30)      { ai = 0; r2 = row; }
    else if (row < 62) { ai = 1; r2 = row - 30; }
    else if (row < 94) { ai = 3; r2 = row - 62; }
    else               { ai = 2; r2 = row - 94; }
    s_concat[row][c] = (_Float16)s_acc[ai][r2][c];
  }
  __syncthreads();

  // K/V of 126-row concat: dual WMMA, one 16-row chunk per wave
  wave_proj_t<true>(&s_concat[0][0], oin_w + 64, oin_b + 8, oin_w + 128, oin_b + 16,
                    &s_kpTf[0][0], &s_vpTf[0][0], 128, lane, wave, wave + 1);
  __syncthreads();

  if (wave == 0) {
    float* scr = s_scr[0];
    wave_proj_t<false>(&s_t[0][0], oin_w, oin_b, nullptr, nullptr,
                       scr, nullptr, 32, lane, 0, 2);
    attn_rows(lane, scr, &s_kpTf[0][0], &s_vpTf[0][0], 128, 126, &s_ob[0][0][0]);
    wave_proj_rm(&s_ob[0][0][0], oout_w, oout_b, &s_tout[0][0], lane, 0, 2);
  }
  __syncthreads();

  // ---------------- Phase 3: fc1 + channel softmax -> (B,3,30) -------------
  if (tid < 90) {
    float sum = fc1_b[tid];
    const float* wr = fc1_w + tid * 256;
    const float* x  = (const float*)s_tout;
    #pragma unroll 8
    for (int j = 0; j < 256; ++j) sum += wr[j] * x[j];
    s_fc[tid] = sum;
  }
  __syncthreads();
  if (tid < 30) {
    float x0 = s_fc[tid * 3 + 0], x1 = s_fc[tid * 3 + 1], x2 = s_fc[tid * 3 + 2];
    float m  = fmaxf(x0, fmaxf(x1, x2));
    float e0 = __expf(x0 - m), e1 = __expf(x1 - m), e2 = __expf(x2 - m);
    float s  = e0 + e1 + e2;
    float* o = out + (size_t)b * 90;
    o[0 * 30 + tid] = e0 / s;
    o[1 * 30 + tid] = e1 / s;
    o[2 * 30 + tid] = e2 / s;
  }
}

extern "C" void kernel_launch(void* const* d_in, const int* in_sizes, int n_in,
                              void* d_out, int out_size, void* d_ws, size_t ws_size,
                              hipStream_t stream) {
  (void)n_in; (void)out_size; (void)d_ws; (void)ws_size;
  const int B = in_sizes[0] / (2 * 20 * 118);
  cmt_kernel<<<B, 256, 0, stream>>>(
      (const float*)d_in[0],  (const float*)d_in[1],  (const float*)d_in[2],
      (const float*)d_in[3],  (const float*)d_in[4],  (const float*)d_in[5],
      (const float*)d_in[6],  (const float*)d_in[7],  (const float*)d_in[8],
      (const float*)d_in[9],  (const float*)d_in[10], (const float*)d_in[11],
      (const float*)d_in[12], (const float*)d_in[13], (const float*)d_in[14],
      (const float*)d_in[15], (const float*)d_in[16], (const float*)d_in[17],
      (const float*)d_in[18], (const float*)d_in[19], (const float*)d_in[20],
      (const float*)d_in[21], (const float*)d_in[22], (const float*)d_in[23],
      (const float*)d_in[24], (const float*)d_in[25], (const float*)d_in[26],
      (const float*)d_in[27], (const float*)d_in[28], (const float*)d_in[29],
      (float*)d_out);
}
